// DEEP_GATGNN_52965536694623
// MI455X (gfx1250) — matrix operand order repaired and checked by reading, not compiled, not run on hardware
//
#include <hip/hip_runtime.h>
#include <hip/hip_bf16.h>

#define N_NODES 8000
#define E_EDGES 128000
#define BGRAPH  64
#define DIM     64
#define HEADS   4
#define GROUPS  10
#define F_NODE  100
#define F_EDGE  50
#define F_GLOB  108
#define N_LAYERS 5
#define LAMDA   0.01f
#define EPSV    1e-5f

typedef __attribute__((ext_vector_type(16))) _Float16 v16h;
typedef __attribute__((ext_vector_type(8)))  _Float16 v8h;
typedef __attribute__((ext_vector_type(8)))  float    v8f;

__device__ __forceinline__ float softplusf(float x) {
    return x > 20.f ? x : log1pf(expf(x));
}
__device__ __forceinline__ float leakyf(float x) { return x > 0.f ? x : 0.2f * x; }

// ---------------- generic helpers ----------------
__global__ void k_zero(float* p, size_t n) {
    for (size_t i = (size_t)blockIdx.x * blockDim.x + threadIdx.x; i < n;
         i += (size_t)gridDim.x * blockDim.x)
        p[i] = 0.f;
}

// Swizzle W [128,256] f32 into per-lane WMMA B-fragment layout (f16):
// Wswz[(((nt*4 + ks)*32 + lane)*16 + t] = W[k*256 + n]
//   with half = lane>>4, n = nt*16 + (lane&15), k = ks*32 + (t>>3)*16 + half*8 + (t&7).
// Each lane's 16 halves (32 B) are contiguous -> one v16h load per fragment.
__global__ void k_swz_w(const float* __restrict__ s, _Float16* __restrict__ d) {
    int i = blockIdx.x * blockDim.x + threadIdx.x;  // over 128*256
    if (i >= 128 * 256) return;
    int t = i & 15;
    int lane = (i >> 4) & 31;
    int ks = (i >> 9) & 3;
    int nt = i >> 11;
    int half = lane >> 4;
    int n = nt * 16 + (lane & 15);
    int k = ks * 32 + (t >> 3) * 16 + half * 8 + (t & 7);
    d[i] = (_Float16)s[k * 256 + n];
}

// ---------------- pre-processing linears ----------------
__global__ void k_pre_node(const float* __restrict__ x, const float* __restrict__ w,
                           const float* __restrict__ b, float* __restrict__ xcur,
                           float* __restrict__ prev) {
    int i = blockIdx.x * blockDim.x + threadIdx.x;  // over N*64
    if (i >= N_NODES * DIM) return;
    int n = i >> 6, d = i & 63;
    float acc = b[d];
    for (int k = 0; k < F_NODE; ++k) acc += x[n * F_NODE + k] * w[k * DIM + d];
    acc = leakyf(acc);
    xcur[i] = acc;
    prev[i] = acc;
}

__global__ void k_pre_edge(const float* __restrict__ ea, const float* __restrict__ w,
                           const float* __restrict__ b, float* __restrict__ oute) {
    int i = blockIdx.x * blockDim.x + threadIdx.x;  // over E*64
    if (i >= E_EDGES * DIM) return;
    int e = i >> 6, d = i & 63;
    float acc = b[d];
    for (int k = 0; k < F_EDGE; ++k) acc += ea[e * F_EDGE + k] * w[k * DIM + d];
    oute[i] = leakyf(acc);
}

// ---------------- WMMA edge GEMM + fused attention logits ----------------
// 16 edges per block. A = [x_row||e] (h_i) and [x_col||e] (h_j), each 16x128 f16 in LDS.
// Wswz: pre-swizzled fragment-major f16 (L2-resident). 8 waves: wave&1 selects i/j
// matrix, wave>>1 selects a group of 4 N-tiles. 4 K-steps of wmma_f32_16x16x32_f16.
__global__ __launch_bounds__(256) void k_edge_gemm(
    const float* __restrict__ xcur, const float* __restrict__ oute,
    const int* __restrict__ row, const int* __restrict__ col,
    const _Float16* __restrict__ Wswz, const float* __restrict__ att,
    _Float16* __restrict__ hj, float* __restrict__ alphaA) {
    __shared__ _Float16 Ai[16][128];
    __shared__ _Float16 Aj[16][128];
    __shared__ float atmp[16 * HEADS];

    const int tid = threadIdx.x;
    const int e0 = blockIdx.x * 16;

    for (int t = tid; t < 2048; t += 256) {
        int m = t >> 7, c = t & 127;
        int e = e0 + m;
        float ev = (c >= 64) ? oute[(size_t)e * 64 + (c - 64)] : 0.f;
        float vi = (c < 64) ? xcur[(size_t)row[e] * 64 + c] : ev;
        float vj = (c < 64) ? xcur[(size_t)col[e] * 64 + c] : ev;
        Ai[m][c] = (_Float16)vi;
        Aj[m][c] = (_Float16)vj;
    }
    if (tid < 64) atmp[tid] = 0.f;
    __syncthreads();

    const int wave = tid >> 5;
    const int lane = tid & 31;
    const int half = lane >> 4;   // K-half selector
    const int mrow = lane & 15;   // A row
    const int ncol = lane & 15;   // B/C column within tile
    const int sel = wave & 1;     // 0: h_i (attention only), 1: h_j (store + attention)
    const int ngrp = wave >> 1;   // 0..3
    const _Float16(*A)[128] = sel ? Aj : Ai;

#pragma unroll
    for (int ntl = 0; ntl < 4; ++ntl) {
        const int nt = ngrp * 4 + ntl;
        const int n0 = nt * 16;
        v8f acc = {};
#pragma unroll
        for (int ks = 0; ks < 4; ++ks) {
            const int k0 = ks * 32;
            // A fragment: two aligned 16-byte LDS loads
            v8h alo = *(const v8h*)&A[mrow][k0 + half * 8];
            v8h ahi = *(const v8h*)&A[mrow][k0 + 16 + half * 8];
            v16h a = __builtin_shufflevector(alo, ahi, 0, 1, 2, 3, 4, 5, 6, 7, 8, 9,
                                             10, 11, 12, 13, 14, 15);
            // B fragment: one contiguous 32-byte load from swizzled W
            v16h b = *(const v16h*)&Wswz[((size_t)(nt * 4 + ks) * 32 + lane) * 16];
            acc = __builtin_amdgcn_wmma_f32_16x16x32_f16(
                false, a, false, b, (short)0, acc, false, false);
        }
        // C layout: VGPR r -> M = half*8 + r, N = n0 + ncol
#pragma unroll
        for (int r = 0; r < 8; ++r) {
            int m = half * 8 + r;
            int n = n0 + ncol;
            float val = softplusf(acc[r]);
            int head = n >> 6, d = n & 63;
            atomicAdd(&atmp[m * HEADS + head], val * att[head * 128 + sel * 64 + d]);
            if (sel) hj[(size_t)(e0 + m) * 256 + n] = (_Float16)val;
        }
    }
    __syncthreads();
    if (tid < 64) {
        int m = tid >> 2, h = tid & 3;
        alphaA[(size_t)(e0 + m) * HEADS + h] = softplusf(atmp[tid]);
    }
}

// ---------------- alpha batch-norm statistics ----------------
__global__ void k_alpha_stats(const float* __restrict__ alphaA, float* __restrict__ stats) {
    __shared__ float ls[8];
    if (threadIdx.x < 8) ls[threadIdx.x] = 0.f;
    __syncthreads();
    float s[HEADS] = {0, 0, 0, 0}, q[HEADS] = {0, 0, 0, 0};
    for (int e = blockIdx.x * blockDim.x + threadIdx.x; e < E_EDGES;
         e += gridDim.x * blockDim.x) {
#pragma unroll
        for (int h = 0; h < HEADS; ++h) {
            float a = alphaA[(size_t)e * HEADS + h];
            s[h] += a;
            q[h] += a * a;
        }
    }
#pragma unroll
    for (int h = 0; h < HEADS; ++h) {
        atomicAdd(&ls[h], s[h]);
        atomicAdd(&ls[4 + h], q[h]);
    }
    __syncthreads();
    if (threadIdx.x < 8) atomicAdd(&stats[threadIdx.x], ls[threadIdx.x]);
}

// BN + softplus, then segment max (softplus > 0 so uint bit order == float order)
__global__ void k_alpha_bn_segmax(const float* __restrict__ alphaA, float* __restrict__ alphaB,
                                  const float* __restrict__ stats, const float* __restrict__ bng,
                                  const float* __restrict__ bnb, const int* __restrict__ row,
                                  unsigned int* __restrict__ segmax) {
    int i = blockIdx.x * blockDim.x + threadIdx.x;
    if (i >= E_EDGES * HEADS) return;
    int e = i >> 2, h = i & 3;
    float mean = stats[h] * (1.f / E_EDGES);
    float var = stats[4 + h] * (1.f / E_EDGES) - mean * mean;
    float a = (alphaA[i] - mean) * rsqrtf(var + EPSV) * bng[h] + bnb[h];
    a = softplusf(a);
    alphaB[i] = a;
    atomicMax(&segmax[(size_t)row[e] * HEADS + h], __float_as_uint(a));
}

__global__ void k_alpha_exp_segsum(float* __restrict__ alphaB,
                                   const unsigned int* __restrict__ segmax,
                                   const int* __restrict__ row, float* __restrict__ segsum) {
    int i = blockIdx.x * blockDim.x + threadIdx.x;
    if (i >= E_EDGES * HEADS) return;
    int e = i >> 2, h = i & 3;
    float m = __uint_as_float(segmax[(size_t)row[e] * HEADS + h]);
    float ea = expf(alphaB[i] - m);
    alphaB[i] = ea;
    atomicAdd(&segsum[(size_t)row[e] * HEADS + h], ea);
}

// ---------------- weighted scatter-add aggregation ----------------
__global__ void k_aggregate(const _Float16* __restrict__ hj, const float* __restrict__ alphaB,
                            const float* __restrict__ segsum, const int* __restrict__ row,
                            float* __restrict__ aggr) {
    size_t total = (size_t)E_EDGES * 256;
    for (size_t i = (size_t)blockIdx.x * blockDim.x + threadIdx.x; i < total;
         i += (size_t)gridDim.x * blockDim.x) {
        int e = (int)(i >> 8), c = (int)(i & 255);
        int h = c >> 6;
        int r = row[e];
        float w = alphaB[(size_t)e * HEADS + h] /
                  (segsum[(size_t)r * HEADS + h] + 1e-16f);
        atomicAdd(&aggr[(size_t)r * 256 + c], (float)hj[i] * w);
    }
}

__global__ void k_node_update(const float* __restrict__ aggr, const float* __restrict__ bias,
                              float* __restrict__ xnew) {
    int i = blockIdx.x * blockDim.x + threadIdx.x;  // over N*64
    if (i >= N_NODES * DIM) return;
    int n = i >> 6, d = i & 63;
    const float* a = &aggr[(size_t)n * 256];
    xnew[i] = 0.25f * (a[d] + a[64 + d] + a[128 + d] + a[192 + d]) + bias[d];
}

// ---------------- DiffGroupNorm ----------------
__global__ void k_gn_softmax(const float* __restrict__ xnew, const float* __restrict__ lw,
                             const float* __restrict__ lb, float* __restrict__ sgn) {
    int n = blockIdx.x * blockDim.x + threadIdx.x;
    if (n >= N_NODES) return;
    float lg[GROUPS];
    float mx = -1e30f;
    for (int g = 0; g < GROUPS; ++g) {
        float acc = lb[g];
        for (int d = 0; d < DIM; ++d) acc += xnew[(size_t)n * DIM + d] * lw[d * GROUPS + g];
        lg[g] = acc;
        mx = fmaxf(mx, acc);
    }
    float sm = 0.f;
    for (int g = 0; g < GROUPS; ++g) {
        lg[g] = expf(lg[g] - mx);
        sm += lg[g];
    }
    float inv = 1.f / sm;
    for (int g = 0; g < GROUPS; ++g) sgn[(size_t)n * GROUPS + g] = lg[g] * inv;
}

// one block per channel c = g*64+d; reduce sum and sumsq over nodes
__global__ __launch_bounds__(256) void k_gn_stats(const float* __restrict__ sgn,
                                                  const float* __restrict__ xnew,
                                                  float* __restrict__ st) {
    __shared__ float rs[256], rq[256];
    int c = blockIdx.x;
    int g = c / DIM, d = c % DIM;
    float s = 0.f, q = 0.f;
    for (int n = threadIdx.x; n < N_NODES; n += 256) {
        float t = sgn[(size_t)n * GROUPS + g] * xnew[(size_t)n * DIM + d];
        s += t;
        q += t * t;
    }
    rs[threadIdx.x] = s;
    rq[threadIdx.x] = q;
    __syncthreads();
    for (int k = 128; k > 0; k >>= 1) {
        if (threadIdx.x < k) {
            rs[threadIdx.x] += rs[threadIdx.x + k];
            rq[threadIdx.x] += rq[threadIdx.x + k];
        }
        __syncthreads();
    }
    if (threadIdx.x == 0) {
        st[c] = rs[0];
        st[GROUPS * DIM + c] = rq[0];
    }
}

__global__ void k_gn_finish(const float* __restrict__ xnew, const float* __restrict__ sgn,
                            const float* __restrict__ st, const float* __restrict__ gamma,
                            const float* __restrict__ beta, float* __restrict__ prev,
                            float* __restrict__ xcur) {
    int i = blockIdx.x * blockDim.x + threadIdx.x;  // over N*64
    if (i >= N_NODES * DIM) return;
    int n = i >> 6, d = i & 63;
    float xv = xnew[i];
    float acc = 0.f;
    for (int g = 0; g < GROUPS; ++g) {
        int c = g * DIM + d;
        float t = sgn[(size_t)n * GROUPS + g] * xv;
        float mean = st[c] * (1.f / N_NODES);
        float var = st[GROUPS * DIM + c] * (1.f / N_NODES) - mean * mean;
        acc += (t - mean) * rsqrtf(var + EPSV) * gamma[c] + beta[c];
    }
    float out = xv + LAMDA * acc + prev[i];
    xcur[i] = out;
    prev[i] = out;
}

// ---------------- global attention pooling head ----------------
__global__ void k_glob0(const float* __restrict__ xcur, const float* __restrict__ gf,
                        const float* __restrict__ w, const float* __restrict__ b,
                        float* __restrict__ h0) {
    int i = blockIdx.x * blockDim.x + threadIdx.x;
    if (i >= N_NODES * DIM) return;
    int n = i >> 6, d = i & 63;
    float acc = b[d];
    for (int k = 0; k < DIM; ++k) acc += xcur[(size_t)n * DIM + k] * w[k * DIM + d];
    for (int k = 0; k < F_GLOB; ++k)
        acc += gf[(size_t)n * F_GLOB + k] * w[(DIM + k) * DIM + d];
    h0[i] = softplusf(acc);
}

__global__ void k_glob1(const float* __restrict__ h0, const float* __restrict__ w,
                        const float* __restrict__ b, float* __restrict__ h1) {
    int i = blockIdx.x * blockDim.x + threadIdx.x;
    if (i >= N_NODES * DIM) return;
    int n = i >> 6, d = i & 63;
    float acc = b[d];
    for (int k = 0; k < DIM; ++k) acc += h0[(size_t)n * DIM + k] * w[k * DIM + d];
    h1[i] = softplusf(acc);
}

__global__ void k_glob2(const float* __restrict__ h1, const float* __restrict__ w,
                        const float* __restrict__ b, float* __restrict__ hsc) {
    int n = blockIdx.x * blockDim.x + threadIdx.x;
    if (n >= N_NODES) return;
    float acc = b[0];
    for (int k = 0; k < DIM; ++k) acc += h1[(size_t)n * DIM + k] * w[k];
    hsc[n] = acc;
}

// one block per graph; batch[] is sorted so we just filter
__global__ __launch_bounds__(256) void k_pool(const float* __restrict__ hsc,
                                              const float* __restrict__ xcur,
                                              const int* __restrict__ batch,
                                              float* __restrict__ pooled) {
    __shared__ float red[256];
    int b = blockIdx.x, tid = threadIdx.x;
    float mx = -1e30f;
    for (int n = tid; n < N_NODES; n += 256)
        if (batch[n] == b) mx = fmaxf(mx, hsc[n]);
    red[tid] = mx;
    __syncthreads();
    for (int k = 128; k > 0; k >>= 1) {
        if (tid < k) red[tid] = fmaxf(red[tid], red[tid + k]);
        __syncthreads();
    }
    float gmax = red[0];
    __syncthreads();
    float sm = 0.f;
    for (int n = tid; n < N_NODES; n += 256)
        if (batch[n] == b) sm += expf(hsc[n] - gmax);
    red[tid] = sm;
    __syncthreads();
    for (int k = 128; k > 0; k >>= 1) {
        if (tid < k) red[tid] += red[tid + k];
        __syncthreads();
    }
    float gsum = red[0] + 1e-16f;
    __syncthreads();
    int d = tid & 63, grp = tid >> 6;  // 4 node-strided groups per feature
    float acc = 0.f;
    for (int n = grp; n < N_NODES; n += 4)
        if (batch[n] == b) acc += xcur[(size_t)n * DIM + d] * (expf(hsc[n] - gmax) / gsum);
    red[tid] = acc;
    __syncthreads();
    if (grp == 0)
        pooled[b * DIM + d] = red[d] + red[64 + d] + red[128 + d] + red[192 + d];
}

__global__ void k_post(const float* __restrict__ pooled, const float* __restrict__ pw,
                       const float* __restrict__ pb, const float* __restrict__ ow,
                       const float* __restrict__ ob, float* __restrict__ out) {
    int b = threadIdx.x;
    if (b >= BGRAPH) return;
    float acc = ob[0];
    for (int k = 0; k < DIM; ++k) {
        float h = pb[k];
        for (int j = 0; j < DIM; ++j) h += pooled[b * DIM + j] * pw[j * DIM + k];
        acc += softplusf(h) * ow[k];
    }
    out[b] = acc;
}

// ---------------- host orchestration ----------------
extern "C" void kernel_launch(void* const* d_in, const int* in_sizes, int n_in,
                              void* d_out, int out_size, void* d_ws, size_t ws_size,
                              hipStream_t stream) {
    (void)in_sizes; (void)n_in; (void)out_size; (void)ws_size;
    // Input order: top-level dict insertion order {x, edge_index, edge_attr, batch,
    // glob_feat}, then params pytree flattened with sorted dict keys:
    // convs(5x{W,att,bias,bn_b,bn_g}), dgns(5x{b,g,lin_b,lin_w}),
    // glob{b0,b1,b2,w0,w1,w2}, out{b,w}, post{b,w}, pre_E{b,w}, pre_N{b,w}
    const float* x_in = (const float*)d_in[0];
    const int* ei = (const int*)d_in[1];
    const float* eattr = (const float*)d_in[2];
    const int* batch = (const int*)d_in[3];
    const float* gfeat = (const float*)d_in[4];
    const int* row = ei;
    const int* col = ei + E_EDGES;

    const float *convW[N_LAYERS], *convAtt[N_LAYERS], *convBias[N_LAYERS],
        *convBnB[N_LAYERS], *convBnG[N_LAYERS];
    for (int l = 0; l < N_LAYERS; ++l) {
        convW[l] = (const float*)d_in[5 + 5 * l + 0];
        convAtt[l] = (const float*)d_in[5 + 5 * l + 1];
        convBias[l] = (const float*)d_in[5 + 5 * l + 2];
        convBnB[l] = (const float*)d_in[5 + 5 * l + 3];
        convBnG[l] = (const float*)d_in[5 + 5 * l + 4];
    }
    const float *dgnB[N_LAYERS], *dgnG[N_LAYERS], *dgnLinB[N_LAYERS], *dgnLinW[N_LAYERS];
    for (int l = 0; l < N_LAYERS; ++l) {
        dgnB[l] = (const float*)d_in[30 + 4 * l + 0];
        dgnG[l] = (const float*)d_in[30 + 4 * l + 1];
        dgnLinB[l] = (const float*)d_in[30 + 4 * l + 2];
        dgnLinW[l] = (const float*)d_in[30 + 4 * l + 3];
    }
    const float* gb0 = (const float*)d_in[50];
    const float* gb1 = (const float*)d_in[51];
    const float* gb2 = (const float*)d_in[52];
    const float* gw0 = (const float*)d_in[53];
    const float* gw1 = (const float*)d_in[54];
    const float* gw2 = (const float*)d_in[55];
    const float* outB = (const float*)d_in[56];
    const float* outW = (const float*)d_in[57];
    const float* postB = (const float*)d_in[58];
    const float* postW = (const float*)d_in[59];
    const float* preEB = (const float*)d_in[60];
    const float* preEW = (const float*)d_in[61];
    const float* preNB = (const float*)d_in[62];
    const float* preNW = (const float*)d_in[63];

    // workspace carve-out
    char* base = (char*)d_ws;
    size_t off = 0;
    auto take = [&](size_t bytes) -> char* {
        char* p = base + off;
        off = (off + bytes + 255) & ~(size_t)255;
        return p;
    };
    _Float16* wswz = (_Float16*)take((size_t)N_LAYERS * 128 * 256 * 2);
    float* out_e = (float*)take((size_t)E_EDGES * DIM * 4);
    float* x_cur = (float*)take((size_t)N_NODES * DIM * 4);
    float* prev = (float*)take((size_t)N_NODES * DIM * 4);
    float* x_new = (float*)take((size_t)N_NODES * DIM * 4);
    float* s_gn = (float*)take((size_t)N_NODES * GROUPS * 4);
    _Float16* h_j = (_Float16*)take((size_t)E_EDGES * 256 * 2);
    float* alphaA = (float*)take((size_t)E_EDGES * HEADS * 4);
    float* alphaB = (float*)take((size_t)E_EDGES * HEADS * 4);
    unsigned int* segmax = (unsigned int*)take((size_t)N_NODES * HEADS * 4);
    float* segsum = (float*)take((size_t)N_NODES * HEADS * 4);
    float* aggr = (float*)take((size_t)N_NODES * 256 * 4);
    float* stats = (float*)take((size_t)(8 + 2 * GROUPS * DIM) * 4);
    float* h0 = (float*)take((size_t)N_NODES * DIM * 4);
    float* h1 = (float*)take((size_t)N_NODES * DIM * 4);
    float* hsc = (float*)take((size_t)N_NODES * 4);
    float* pooled = (float*)take((size_t)BGRAPH * DIM * 4);

    const int TPB = 256;
    const int gND = (N_NODES * DIM + TPB - 1) / TPB;    // 2000
    const int gED = (E_EDGES * DIM + TPB - 1) / TPB;    // 32000
    const int gEH = (E_EDGES * HEADS + TPB - 1) / TPB;  // 2000
    const int gN = (N_NODES + TPB - 1) / TPB;           // 32

    for (int l = 0; l < N_LAYERS; ++l)
        k_swz_w<<<(128 * 256 + TPB - 1) / TPB, TPB, 0, stream>>>(
            convW[l], wswz + (size_t)l * 128 * 256);

    k_pre_node<<<gND, TPB, 0, stream>>>(x_in, preNW, preNB, x_cur, prev);
    k_pre_edge<<<gED, TPB, 0, stream>>>(eattr, preEW, preEB, out_e);

    for (int l = 0; l < N_LAYERS; ++l) {
        k_zero<<<8, TPB, 0, stream>>>(stats, 8 + 2 * GROUPS * DIM);
        k_zero<<<128, TPB, 0, stream>>>((float*)segmax, (size_t)N_NODES * HEADS);
        k_zero<<<128, TPB, 0, stream>>>(segsum, (size_t)N_NODES * HEADS);
        k_zero<<<2048, TPB, 0, stream>>>(aggr, (size_t)N_NODES * 256);

        k_edge_gemm<<<E_EDGES / 16, TPB, 0, stream>>>(
            x_cur, out_e, row, col, wswz + (size_t)l * 128 * 256, convAtt[l], h_j, alphaA);
        k_alpha_stats<<<512, TPB, 0, stream>>>(alphaA, stats);
        k_alpha_bn_segmax<<<gEH, TPB, 0, stream>>>(alphaA, alphaB, stats, convBnG[l],
                                                   convBnB[l], row, segmax);
        k_alpha_exp_segsum<<<gEH, TPB, 0, stream>>>(alphaB, segmax, row, segsum);
        k_aggregate<<<16384, TPB, 0, stream>>>(h_j, alphaB, segsum, row, aggr);
        k_node_update<<<gND, TPB, 0, stream>>>(aggr, convBias[l], x_new);

        k_gn_softmax<<<gN, TPB, 0, stream>>>(x_new, dgnLinW[l], dgnLinB[l], s_gn);
        k_gn_stats<<<GROUPS * DIM, TPB, 0, stream>>>(s_gn, x_new, stats + 8);
        k_gn_finish<<<gND, TPB, 0, stream>>>(x_new, s_gn, stats + 8, dgnG[l], dgnB[l],
                                             prev, x_cur);
    }

    k_glob0<<<gND, TPB, 0, stream>>>(x_cur, gfeat, gw0, gb0, h0);
    k_glob1<<<gND, TPB, 0, stream>>>(h0, gw1, gb1, h1);
    k_glob2<<<gN, TPB, 0, stream>>>(h1, gw2, gb2, hsc);
    k_pool<<<BGRAPH, TPB, 0, stream>>>(hsc, x_cur, batch, pooled);
    k_post<<<1, 64, 0, stream>>>(pooled, postW, postB, outW, outB, (float*)d_out);
}
